// _PropAttnWrapper_6700148982393
// MI455X (gfx1250) — matrix operand order repaired and checked by reading, not compiled, not run on hardware
//
#include <hip/hip_runtime.h>
#include <hip/hip_bf16.h>

typedef __bf16 bf16_t;
typedef __attribute__((ext_vector_type(16))) __bf16 v16bf;
typedef __attribute__((ext_vector_type(8)))  float  v8f;

#define BHD 64
#define NH 12
#define SEQ 2048
#define CDIM 768

__device__ __forceinline__ unsigned short f2bf_u16(float f) {
    unsigned u = __builtin_bit_cast(unsigned, f);
    return (unsigned short)((u + 0x7FFFu + ((u >> 16) & 1u)) >> 16);
}
__device__ __forceinline__ bf16_t f2bf(float f) {
    unsigned short s = f2bf_u16(f);
    return __builtin_bit_cast(bf16_t, s);
}
__device__ __forceinline__ unsigned pk2(float a, float b) {
    return (unsigned)f2bf_u16(a) | ((unsigned)f2bf_u16(b) << 16);
}

__device__ __forceinline__ v8f wmma_bf16(v16bf a, v16bf b, v8f c) {
    return __builtin_amdgcn_wmma_f32_16x16x32_bf16(
        false, a, false, b, (short)0, c, false, false);
}

// Async global->LDS 16-byte copy (CDNA5 ASYNCcnt path).
// Low 32 bits of a generic LDS pointer are the LDS byte offset (ISA 10.2).
__device__ __forceinline__ void async_cp16(bf16_t* lds_dst, const bf16_t* gsrc) {
    unsigned lds = (unsigned)(unsigned long long)lds_dst;
    asm volatile("global_load_async_to_lds_b128 %0, %1, off"
                 :: "v"(lds), "v"(gsrc) : "memory");
}
__device__ __forceinline__ void wait_async0() {
    asm volatile("s_wait_asynccnt 0x0" ::: "memory");
}

// A fragment (16x32 bf16, MxK): row-major source, row = lane&15,
// lanes<16 hold K {0..15}, lanes>=16 hold K {16..31}.
__device__ __forceinline__ v16bf frag_a(const bf16_t* base, int row0, int stride, int koff) {
    int lane = threadIdx.x & 31;
    int m = lane & 15, hi = lane >> 4;
    const bf16_t* p = base + (size_t)(row0 + m) * stride + koff + hi * 16;
    v16bf f;
#pragma unroll
    for (int g = 0; g < 2; ++g)
#pragma unroll
        for (int j = 0; j < 8; ++j)
            f[g * 8 + j] = p[g * 8 + j];
    return f;
}

// B fragment (32x16 bf16, KxN) from [n][k] layout (k contiguous).
__device__ __forceinline__ v16bf frag_b_nk(const bf16_t* base, int n0, int stride, int koff) {
    int lane = threadIdx.x & 31;
    int n = lane & 15, hi = lane >> 4;
    const bf16_t* p = base + (size_t)(n0 + n) * stride + koff + hi * 16;
    v16bf f;
#pragma unroll
    for (int j = 0; j < 16; ++j) f[j] = p[j];
    return f;
}

// B fragment from [k][n] layout (n contiguous, strided in k).
__device__ __forceinline__ v16bf frag_b_kn(const bf16_t* base, int n0, int stride, int koff) {
    int lane = threadIdx.x & 31;
    int n = lane & 15, hi = lane >> 4;
    const bf16_t* p = base + (size_t)(koff + hi * 16) * stride + n0 + n;
    v16bf f;
#pragma unroll
    for (int j = 0; j < 16; ++j) f[j] = p[(size_t)j * stride];
    return f;
}

// ---------------------------------------------------------------------------
// Prep kernel A: fp32 -> bf16 elementwise (x), 8 elems/thread, packed stores.
// ---------------------------------------------------------------------------
__global__ __launch_bounds__(256) void conv_kernel(
    const float* __restrict__ src, bf16_t* __restrict__ dst, long total)
{
    long i = ((long)blockIdx.x * 256 + threadIdx.x) * 8;
    if (i >= total) return;
    const float4* s = (const float4*)(src + i);
    float4 f0 = s[0], f1 = s[1];
    uint4 o;
    o.x = pk2(f0.x, f0.y); o.y = pk2(f0.z, f0.w);
    o.z = pk2(f1.x, f1.y); o.w = pk2(f1.z, f1.w);
    *(uint4*)(dst + i) = o;
}

// ---------------------------------------------------------------------------
// Prep kernel B: transpose+convert weights: src[K][N] f32 -> dst[N][K] bf16.
// ---------------------------------------------------------------------------
__global__ __launch_bounds__(256) void transpose_conv_kernel(
    const float* __restrict__ src, bf16_t* __restrict__ dst, int K, int N)
{
    long i = ((long)blockIdx.x * 256 + threadIdx.x) * 2;
    if (i >= (long)K * N) return;
    int n = (int)(i / K);
    int k = (int)(i - (long)n * K);
    unsigned v = pk2(src[(size_t)k * N + n], src[(size_t)(k + 1) * N + n]);
    *(unsigned*)(dst + i) = v;
}

// ---------------------------------------------------------------------------
// Shared GEMM core: 128x128 block, BK=64, double-buffered async LDS staging.
// A (bf16 [M][768]) x B^T (bf16 [N][768]) -> acc[2][4] per wave.
// Dynamic LDS: As = smem[2][128][72], Bs follows.
// ---------------------------------------------------------------------------
#define LDSTRIDE 72
#define TILE_ELEMS (128 * LDSTRIDE)

__device__ __forceinline__ void gemm_core(
    const bf16_t* __restrict__ Abase, const bf16_t* __restrict__ Bbase,
    bf16_t* As, bf16_t* Bs, int bm0, int bn0, v8f acc[2][4])
{
    int tid = threadIdx.x;
    int wid = tid >> 5;
    int wm = wid & 3, wn = wid >> 2;          // 4 x 2 wave grid; 32x64 per wave
    int sr = tid >> 1, sc = (tid & 1) * 32;   // staging: 1 row, 32 k per thread

    // prologue: async-stage k0=0 into buffer 0
    {
        const bf16_t* sa = Abase + (size_t)(bm0 + sr) * CDIM + sc;
        const bf16_t* sb = Bbase + (size_t)(bn0 + sr) * CDIM + sc;
        bf16_t* da = As + (size_t)sr * LDSTRIDE + sc;
        bf16_t* db = Bs + (size_t)sr * LDSTRIDE + sc;
#pragma unroll
        for (int c = 0; c < 4; ++c) {
            async_cp16(da + c * 8, sa + c * 8);
            async_cp16(db + c * 8, sb + c * 8);
        }
    }
    wait_async0();
    __syncthreads();

    int buf = 0;
    for (int k0 = 0; k0 < CDIM; k0 += 64) {
        int nk = k0 + 64;
        if (nk < CDIM) {   // async-stage next K-tile into the other buffer
            int nb = buf ^ 1;
            const bf16_t* sa = Abase + (size_t)(bm0 + sr) * CDIM + nk + sc;
            const bf16_t* sb = Bbase + (size_t)(bn0 + sr) * CDIM + nk + sc;
            bf16_t* da = As + (size_t)nb * TILE_ELEMS + (size_t)sr * LDSTRIDE + sc;
            bf16_t* db = Bs + (size_t)nb * TILE_ELEMS + (size_t)sr * LDSTRIDE + sc;
#pragma unroll
            for (int c = 0; c < 4; ++c) {
                async_cp16(da + c * 8, sa + c * 8);
                async_cp16(db + c * 8, sb + c * 8);
            }
        }
        const bf16_t* Ab = As + (size_t)buf * TILE_ELEMS;
        const bf16_t* Bb = Bs + (size_t)buf * TILE_ELEMS;
#pragma unroll
        for (int d0 = 0; d0 < 2; ++d0) {
            v16bf a[2], b[4];
#pragma unroll
            for (int i = 0; i < 2; ++i) a[i] = frag_a(Ab, wm * 32 + i * 16, LDSTRIDE, d0 * 32);
#pragma unroll
            for (int j = 0; j < 4; ++j) b[j] = frag_b_nk(Bb, wn * 64 + j * 16, LDSTRIDE, d0 * 32);
#pragma unroll
            for (int i = 0; i < 2; ++i)
#pragma unroll
                for (int j = 0; j < 4; ++j)
                    acc[i][j] = wmma_bf16(a[i], b[j], acc[i][j]);
        }
        wait_async0();      // own async loads for next tile done
        __syncthreads();    // everyone's loads done; buf consumable next iter
        buf ^= 1;
    }
}

// ---------------------------------------------------------------------------
// Kernel 1: QKV GEMM  Y[8192,2304] = Xb @ WqkvT^T + b   (bf16 WMMA)
// Scatter into Q/K/V ws [B,H,N,64] bf16; fold 1/sqrt(64) into Q.
// ---------------------------------------------------------------------------
__global__ __launch_bounds__(256) void qkv_gemm_kernel(
    const bf16_t* __restrict__ Xb, const bf16_t* __restrict__ WT,
    const float* __restrict__ bias,
    bf16_t* __restrict__ qws, bf16_t* __restrict__ kws, bf16_t* __restrict__ vws)
{
    extern __shared__ __align__(16) bf16_t smem[];
    bf16_t* As = smem;
    bf16_t* Bs = smem + 2 * TILE_ELEMS;
    int bm0 = blockIdx.y * 128;
    int bn0 = blockIdx.x * 128;
    int lane = threadIdx.x & 31, wid = threadIdx.x >> 5;
    int wm = wid & 3, wn = wid >> 2;
    int hi = lane >> 4, ln = lane & 15;

    v8f acc[2][4];
#pragma unroll
    for (int i = 0; i < 2; ++i)
#pragma unroll
        for (int j = 0; j < 4; ++j)
#pragma unroll
            for (int e = 0; e < 8; ++e) acc[i][j][e] = 0.0f;

    gemm_core(Xb, WT, As, Bs, bm0, bn0, acc);

#pragma unroll
    for (int i = 0; i < 2; ++i)
#pragma unroll
        for (int j = 0; j < 4; ++j) {
            int nn = bn0 + wn * 64 + j * 16 + ln;
            float bv = bias[nn];
            int which = nn / CDIM;
            int rem = nn - which * CDIM;
            int h = rem >> 6, hd = rem & 63;
            bf16_t* dstb = (which == 0) ? qws : (which == 1) ? kws : vws;
            float scl = (which == 0) ? 0.125f : 1.0f;
#pragma unroll
            for (int r = 0; r < 8; ++r) {
                int mm = bm0 + wm * 32 + i * 16 + hi * 8 + r;
                int bb = mm >> 11, q = mm & 2047;
                float v = (acc[i][j][r] + bv) * scl;
                dstb[((size_t)(bb * NH + h) * SEQ + q) * BHD + hd] = f2bf(v);
            }
        }
}

// ---------------------------------------------------------------------------
// Kernel 2: flash attention per (b,h). 128 q-rows/block, 16 rows per wave,
// streaming 64-key blocks, double-buffered async K/V staging, online softmax
// with key-side log bias.
// ---------------------------------------------------------------------------
__global__ __launch_bounds__(256) void attn_kernel(
    const bf16_t* __restrict__ qws, const bf16_t* __restrict__ kws,
    const bf16_t* __restrict__ vws, const float* __restrict__ klb,
    bf16_t* __restrict__ attn_out)
{
    __shared__ __align__(16) bf16_t Kt[2][64][72];   // [buf][key][d]
    __shared__ __align__(16) bf16_t Vt[2][64][72];   // [buf][key][d]
    __shared__ __align__(16) bf16_t Pt[128][72];     // per-wave 16 rows x 64 keys

    int blk = blockIdx.x;
    const int nq = SEQ / 128;                 // 16 q-blocks per head
    int bh = blk / nq;
    int qblk = blk - bh * nq;
    int b = bh / NH, h = bh - b * NH;
    int q0 = qblk * 128;
    int tid = threadIdx.x, wid = tid >> 5, lane = tid & 31;
    int hi = lane >> 4, ln = lane & 15;

    const bf16_t* Qb = qws + (size_t)bh * SEQ * BHD;
    const bf16_t* Kb = kws + (size_t)bh * SEQ * BHD;
    const bf16_t* Vb = vws + (size_t)bh * SEQ * BHD;
    const float* biasrow = klb + (size_t)b * SEQ;

    // Q fragments for this wave's 16 rows, kept in registers all kernel.
    int qrow = q0 + wid * 16;
    v16bf qa0 = frag_a(Qb + (size_t)qrow * BHD, 0, BHD, 0);
    v16bf qa1 = frag_a(Qb + (size_t)qrow * BHD, 0, BHD, 32);

    float rmax[8], rsum[8];
    v8f oacc[4];
#pragma unroll
    for (int r = 0; r < 8; ++r) { rmax[r] = -1e30f; rsum[r] = 0.0f; }
#pragma unroll
    for (int dt = 0; dt < 4; ++dt)
#pragma unroll
        for (int e = 0; e < 8; ++e) oacc[dt][e] = 0.0f;

    int kr = tid >> 2, c0 = (tid & 3) * 16;   // staging map: row, 16 d per thread
    // prologue: stage key-block 0 into buffer 0
    {
        const bf16_t* ks = Kb + (size_t)kr * BHD + c0;
        const bf16_t* vs = Vb + (size_t)kr * BHD + c0;
        async_cp16(&Kt[0][kr][c0], ks);
        async_cp16(&Kt[0][kr][c0 + 8], ks + 8);
        async_cp16(&Vt[0][kr][c0], vs);
        async_cp16(&Vt[0][kr][c0 + 8], vs + 8);
    }
    wait_async0();
    __syncthreads();

    int buf = 0;
    for (int kb = 0; kb < SEQ; kb += 64) {
        if (kb + 64 < SEQ) {   // async-stage next key block into other buffer
            int nb = buf ^ 1;
            const bf16_t* ks = Kb + (size_t)(kb + 64 + kr) * BHD + c0;
            const bf16_t* vs = Vb + (size_t)(kb + 64 + kr) * BHD + c0;
            async_cp16(&Kt[nb][kr][c0], ks);
            async_cp16(&Kt[nb][kr][c0 + 8], ks + 8);
            async_cp16(&Vt[nb][kr][c0], vs);
            async_cp16(&Vt[nb][kr][c0 + 8], vs + 8);
        }

        // S = Q @ K^T  (16 x 64 per wave)
        v8f s[4];
#pragma unroll
        for (int nt = 0; nt < 4; ++nt)
#pragma unroll
            for (int e = 0; e < 8; ++e) s[nt][e] = 0.0f;
#pragma unroll
        for (int d0 = 0; d0 < 2; ++d0) {
            v16bf aq = d0 ? qa1 : qa0;
#pragma unroll
            for (int nt = 0; nt < 4; ++nt) {
                v16bf bk = frag_b_nk(&Kt[buf][0][0], nt * 16, 72, d0 * 32);
                s[nt] = wmma_bf16(aq, bk, s[nt]);
            }
        }

        // key-side bias (per column)
        float cb[4];
#pragma unroll
        for (int nt = 0; nt < 4; ++nt) cb[nt] = biasrow[kb + nt * 16 + ln];
#pragma unroll
        for (int nt = 0; nt < 4; ++nt)
#pragma unroll
            for (int r = 0; r < 8; ++r) s[nt][r] += cb[nt];

        // online softmax per row (rows hi*8+r of this wave's strip)
#pragma unroll
        for (int r = 0; r < 8; ++r) {
            float mt = s[0][r];
#pragma unroll
            for (int nt = 1; nt < 4; ++nt) mt = fmaxf(mt, s[nt][r]);
#pragma unroll
            for (int mk = 1; mk < 16; mk <<= 1) mt = fmaxf(mt, __shfl_xor(mt, mk, 32));
            float mnew = fmaxf(rmax[r], mt);
            float alpha = __expf(rmax[r] - mnew);
            rmax[r] = mnew;
            float ps = 0.0f;
            int prow = wid * 16 + hi * 8 + r;
#pragma unroll
            for (int nt = 0; nt < 4; ++nt) {
                float p = __expf(s[nt][r] - mnew);
                ps += p;
                Pt[prow][nt * 16 + ln] = f2bf(p);
            }
#pragma unroll
            for (int mk = 1; mk < 16; mk <<= 1) ps += __shfl_xor(ps, mk, 32);
            rsum[r] = rsum[r] * alpha + ps;
#pragma unroll
            for (int dt = 0; dt < 4; ++dt) oacc[dt][r] *= alpha;
        }

        // O += P @ V  (only this wave reads its own Pt strip — no barrier)
#pragma unroll
        for (int ks2 = 0; ks2 < 2; ++ks2) {
            v16bf pa = frag_a(&Pt[wid * 16][0], 0, 72, ks2 * 32);
#pragma unroll
            for (int dt = 0; dt < 4; ++dt) {
                v16bf bv = frag_b_kn(&Vt[buf][0][0], dt * 16, 72, ks2 * 32);
                oacc[dt] = wmma_bf16(pa, bv, oacc[dt]);
            }
        }

        wait_async0();      // own async loads for next block done
        __syncthreads();    // all waves done with current buf + loads complete
        buf ^= 1;
    }

    // normalize and store to attention output [B*N, C] bf16
#pragma unroll
    for (int r = 0; r < 8; ++r) {
        float inv = 1.0f / rsum[r];
        int q = q0 + wid * 16 + hi * 8 + r;
        bf16_t* dst = attn_out + ((size_t)(b * SEQ + q)) * CDIM + h * BHD;
#pragma unroll
        for (int dt = 0; dt < 4; ++dt)
            dst[dt * 16 + ln] = f2bf(oacc[dt][r] * inv);
    }
}

// ---------------------------------------------------------------------------
// Kernel 3: output projection  out[8192,768] = A @ WpT^T + b (fp32 out)
// ---------------------------------------------------------------------------
__global__ __launch_bounds__(256) void proj_gemm_kernel(
    const bf16_t* __restrict__ A, const bf16_t* __restrict__ WT,
    const float* __restrict__ bias, float* __restrict__ out)
{
    extern __shared__ __align__(16) bf16_t smem[];
    bf16_t* As = smem;
    bf16_t* Bs = smem + 2 * TILE_ELEMS;
    int bm0 = blockIdx.y * 128;
    int bn0 = blockIdx.x * 128;
    int lane = threadIdx.x & 31, wid = threadIdx.x >> 5;
    int wm = wid & 3, wn = wid >> 2;
    int hi = lane >> 4, ln = lane & 15;

    v8f acc[2][4];
#pragma unroll
    for (int i = 0; i < 2; ++i)
#pragma unroll
        for (int j = 0; j < 4; ++j)
#pragma unroll
            for (int e = 0; e < 8; ++e) acc[i][j][e] = 0.0f;

    gemm_core(A, WT, As, Bs, bm0, bn0, acc);

#pragma unroll
    for (int i = 0; i < 2; ++i)
#pragma unroll
        for (int j = 0; j < 4; ++j) {
            int nn = bn0 + wn * 64 + j * 16 + ln;
            float bv = bias[nn];
#pragma unroll
            for (int r = 0; r < 8; ++r) {
                int mm = bm0 + wm * 32 + i * 16 + hi * 8 + r;
                out[(size_t)mm * CDIM + nn] = acc[i][j][r] + bv;
            }
        }
}

extern "C" void kernel_launch(void* const* d_in, const int* in_sizes, int n_in,
                              void* d_out, int out_size, void* d_ws, size_t ws_size,
                              hipStream_t stream) {
    (void)in_sizes; (void)n_in; (void)out_size; (void)ws_size;
    const float* x      = (const float*)d_in[0];
    const float* w_qkv  = (const float*)d_in[1];
    const float* b_qkv  = (const float*)d_in[2];
    const float* w_proj = (const float*)d_in[3];
    const float* b_proj = (const float*)d_in[4];
    const float* klb    = (const float*)d_in[5];
    float* out = (float*)d_out;

    const size_t rows   = (size_t)4 * SEQ;            // 8192
    const size_t per    = rows * CDIM;                // 6,291,456 (= B*H*N*64)
    bf16_t* xb     = (bf16_t*)d_ws;
    bf16_t* wqkvT  = xb + per;                        // [2304][768]
    bf16_t* wprojT = wqkvT + (size_t)3 * CDIM * CDIM; // [768][768]
    bf16_t* qws    = wprojT + (size_t)CDIM * CDIM;
    bf16_t* kws    = qws + per;
    bf16_t* vws    = kws + per;
    bf16_t* attn   = vws + per;

    const size_t gemm_lds = (size_t)4 * TILE_ELEMS * sizeof(bf16_t);  // 73728 B

    dim3 blk(256);
    // prep: convert x; transpose+convert both weight matrices
    conv_kernel<<<dim3((unsigned)(per / (256 * 8))), blk, 0, stream>>>(x, xb, (long)per);
    transpose_conv_kernel<<<dim3((unsigned)((size_t)3 * CDIM * CDIM / 512)), blk, 0, stream>>>(
        w_qkv, wqkvT, CDIM, 3 * CDIM);
    transpose_conv_kernel<<<dim3((unsigned)((size_t)CDIM * CDIM / 512)), blk, 0, stream>>>(
        w_proj, wprojT, CDIM, CDIM);
    // QKV: M=8192 (64 tiles), N=2304 (18 tiles)
    qkv_gemm_kernel<<<dim3(18, 64), blk, gemm_lds, stream>>>(xb, wqkvT, b_qkv, qws, kws, vws);
    // attention: (B*H)=48 heads x 16 q-blocks = 768 blocks
    attn_kernel<<<dim3(768), blk, 0, stream>>>(qws, kws, vws, klb, attn);
    // projection: M=8192 (64 tiles), N=768 (6 tiles)
    proj_gemm_kernel<<<dim3(6, 64), blk, gemm_lds, stream>>>(attn, wprojT, b_proj, out);
}
